// NAML_56779467653481
// MI455X (gfx1250) — compile-verified, matrix operand order checked by reading, not run on hardware
//
#include <hip/hip_runtime.h>
#include <hip/hip_bf16.h>
#include <math.h>

// ---- model dims ----
#define EMB   300
#define NH    20
#define DKK   20
#define DD    400          // NH*DKK
#define DOUT  64
#define BZd   16
#define HIST  50
#define NPR   5
#define TLEN  30
#define BLEN  100
#define NNEWS (BZd*NPR)          // 80
#define NITEMS (BZd*(NPR+NPR==NPR?0:HIST))  // placeholder (redefined below)
#undef NITEMS
#define NITEMS (BZd*(NPR+HIST))  // 880
#define EKS   320                // EMB padded to K-step multiple of 32
#define YS    416                // DD padded to K-step multiple of 32
#define NPAD  (NH*32)            // 640: per-head DK padded 20->32
#define QKV_NT (NPAD/16)         // 40 column tiles in padded QKV weights
#define POOL_NT (DD/16)          // 25 column tiles in pool W1
#define QKV_KB (EKS/32)          // 10 K-blocks
#define POOL_KB (YS/32)          // 13 K-blocks
#define SCALE_ATT 0.22360679775f // 1/sqrt(20)

typedef __attribute__((ext_vector_type(16))) _Float16 v16h;
typedef __attribute__((ext_vector_type(8)))  _Float16 v8h;
typedef __attribute__((ext_vector_type(8)))  float    v8f;

__device__ __forceinline__ v8f wmma_f16(v16h a, v16h b, v8f c) {
  return __builtin_amdgcn_wmma_f32_16x16x32_f16(false, a, false, b, (short)0, c,
                                                false, false);
}

// Row-major fragment (used for A, and for B given a transposed source):
// lane holds row r0+(lane&15); halves are two contiguous 16B runs at
// k0+8g and k0+8g+16  ->  2x ds_load_b128 / global_load_b128, no branches.
__device__ __forceinline__ v16h frag_row(const _Float16* __restrict__ buf,
                                         int stride, int r0, int k0) {
  int lane = threadIdx.x & 31;
  const _Float16* p =
      buf + (size_t)(r0 + (lane & 15)) * stride + k0 + ((lane >> 4) << 3);
  v8h lo = *(const v8h*)p;
  v8h hi = *(const v8h*)(p + 16);
  v16h a;
#pragma unroll
  for (int h = 0; h < 8; ++h) {
    a[h] = lo[h];
    a[h + 8] = hi[h];
  }
  return a;
}

// Pre-swizzled B fragment: tile (kb, nt), contiguous 32B per lane.
__device__ __forceinline__ v16h frag_swz(const _Float16* __restrict__ W,
                                         int ntiles, int kb, int nt) {
  int lane = threadIdx.x & 31;
  return *(const v16h*)(W + ((((size_t)kb * ntiles + nt) * 32 + lane) << 4));
}

// ---- weight preprocessing: f32 -> f16, swizzled to WMMA B-fragment order ----
// QKV: [EMB][DD] -> padded [EKS][NPAD] (per-head 20->32, zeros) in tile order
__global__ void prep_qkv_swz(const float* __restrict__ W,
                             _Float16* __restrict__ out, int n) {
  int idx = blockIdx.x * 256 + threadIdx.x;
  if (idx >= n) return;
  int h = idx & 15;
  int lane = (idx >> 4) & 31;
  int t = idx >> 9;  // kb*QKV_NT + nt
  int nt = t % QKV_NT;
  int kb = t / QKV_NT;
  int k = kb * 32 + (h & 7) + ((lane >> 4) << 3) + ((h >> 3) << 4);
  int npad = nt * 16 + (lane & 15);
  int hd = npad >> 5, nn = npad & 31;
  float v = 0.0f;
  if (k < EMB && nn < DKK) v = W[k * DD + hd * DKK + nn];
  out[idx] = (_Float16)v;
}

// Pool W1: [DD][DD] -> [YS][DD] (K padded with zeros) in tile order
__global__ void prep_pool_swz(const float* __restrict__ W,
                              _Float16* __restrict__ out, int n) {
  int idx = blockIdx.x * 256 + threadIdx.x;
  if (idx >= n) return;
  int h = idx & 15;
  int lane = (idx >> 4) & 31;
  int t = idx >> 9;  // kb*POOL_NT + nt
  int nt = t % POOL_NT;
  int kb = t / POOL_NT;
  int k = kb * 32 + (h & 7) + ((lane >> 4) << 3) + ((h >> 3) << 4);
  int col = nt * 16 + (lane & 15);
  float v = (k < DD) ? W[k * DD + col] : 0.0f;
  out[idx] = (_Float16)v;
}

// bias [DD] -> padded [NPAD] with zeros in per-head pad lanes
__global__ void prep_bias_pad(const float* __restrict__ b,
                              float* __restrict__ out) {
  int i = blockIdx.x * 256 + threadIdx.x;
  if (i < NPAD) {
    int hd = i >> 5, nn = i & 31;
    out[i] = (nn < DKK) ? b[hd * DKK + nn] : 0.0f;
  }
}

// One workgroup per item. Full MHA + additive pool in LDS with WMMA GEMMs.
template <int L, int LP>
__global__ __launch_bounds__(256) void encoder_kernel(
    const int* __restrict__ news_tok, const int* __restrict__ user_tok,
    const float* __restrict__ embed, const _Float16* __restrict__ Wq,
    const _Float16* __restrict__ Wk, const _Float16* __restrict__ Wv,
    const float* __restrict__ bq, const float* __restrict__ bk,
    const float* __restrict__ bv, const _Float16* __restrict__ W1h,
    const float* __restrict__ pB1, const float* __restrict__ pW2,
    float* __restrict__ outp) {
  extern __shared__ char smem[];
  _Float16* xe = (_Float16*)smem;           // [LP][EKS]  token embeddings
  _Float16* y  = xe + LP * EKS;             // [LP][YS]   MHA output (padded K)
  _Float16* qh = y + LP * YS;               // [LP][32]   per-head Q
  _Float16* kh = qh + LP * 32;              // [LP][32]   per-head K
  _Float16* vt = kh + LP * 32;              // [32][LP]   per-head V transposed
  float*    sc = (float*)(vt + 32 * LP);    // [LP][LP]   scores f32
  _Float16* ph = (_Float16*)(sc + LP * LP); // [LP][LP]   probs f16
  float*    s  = (float*)(ph + LP * LP);    // [LP]       pool logits/weights

  const int item = blockIdx.x;
  const int tid = threadIdx.x;
  const int wave = tid >> 5;
  const int lane = tid & 31;
  const int* tok = (item < NNEWS) ? (news_tok + (size_t)item * L)
                                  : (user_tok + (size_t)(item - NNEWS) * L);

  // 1) gather embeddings -> f16 LDS (zero-padded rows/cols); one wave per row
  for (int l = wave; l < LP; l += 8) {
    const bool valid = (l < L);
    const float* er = embed + (size_t)(valid ? tok[l] : 0) * EMB;
    for (int e = lane; e < EKS; e += 32) {
      float v = (valid && e < EMB) ? er[e] : 0.0f;
      xe[l * EKS + e] = (_Float16)v;
    }
  }
  // zero the K-pad columns of y (cols 400..415) once
  for (int idx = tid; idx < LP * (YS - DD); idx += 256) {
    int l = idx / (YS - DD), c = idx - l * (YS - DD);
    y[l * YS + DD + c] = (_Float16)0.0f;
  }
  __syncthreads();

  const int MT = LP / 16;
  for (int hd = 0; hd < NH; ++hd) {
    // --- QKV projection for this head: [LP,EKS] x [EKS,32] ---
    const int njobs = MT * 2 * 3;
    for (int job = wave; job < njobs; job += 8) {
      int p = job % 3;
      int rem = job / 3;
      int nt = rem & 1;
      int mt = rem >> 1;
      const _Float16* W = (p == 0) ? Wq : (p == 1 ? Wk : Wv);
      const float* bias = (p == 0) ? bq : (p == 1 ? bk : bv);
      v8f acc = {};
      for (int kb = 0; kb < QKV_KB; ++kb) {
        v16h a = frag_row(xe, EKS, mt * 16, kb * 32);
        v16h b = frag_swz(W, QKV_NT, kb, hd * 2 + nt);
        acc = wmma_f16(a, b, acc);
      }
      int nn = nt * 16 + (lane & 15);          // 0..31 within head (pad -> 0)
      int g = lane >> 4;
      float bv2 = bias[hd * 32 + nn];          // zero in pad lanes
#pragma unroll
      for (int j = 0; j < 8; ++j) {
        int m = mt * 16 + j + g * 8;
        float val = acc[j] + bv2;
        if (p == 0)      qh[m * 32 + nn] = (_Float16)val;
        else if (p == 1) kh[m * 32 + nn] = (_Float16)val;
        else             vt[nn * LP + m] = (_Float16)val;
      }
    }
    __syncthreads();

    // --- scores = Q K^T (padded K-cols are exact zeros) ---
    for (int job = wave; job < MT * MT; job += 8) {
      int mt = job / MT, nt = job - mt * MT;
      v16h a = frag_row(qh, 32, mt * 16, 0);
      v16h b = frag_row(kh, 32, nt * 16, 0);  // B^T source: row-major path
      v8f acc = {};
      acc = wmma_f16(a, b, acc);
      int n = nt * 16 + (lane & 15);
      int g = lane >> 4;
#pragma unroll
      for (int j = 0; j < 8; ++j) sc[(mt * 16 + j + g * 8) * LP + n] = acc[j];
    }
    __syncthreads();

    // --- masked softmax per row -> f16 probs ---
    for (int l = tid; l < LP; l += 256) {
      if (l < L) {
        float mx = -1e30f;
        for (int m = 0; m < L; ++m) mx = fmaxf(mx, sc[l * LP + m]);
        float sum = 0.0f;
        for (int m = 0; m < L; ++m) {
          float e = __expf((sc[l * LP + m] - mx) * SCALE_ATT);
          sc[l * LP + m] = e;
          sum += e;
        }
        float inv = 1.0f / sum;
        for (int m = 0; m < LP; ++m)
          ph[l * LP + m] = (m < L) ? (_Float16)(sc[l * LP + m] * inv)
                                   : (_Float16)0.0f;
      } else {
        for (int m = 0; m < LP; ++m) ph[l * LP + m] = (_Float16)0.0f;
      }
    }
    __syncthreads();

    // --- y[:, head cols] = probs @ V  (V^T gives row-major B path) ---
    for (int job = wave; job < MT * 2; job += 8) {
      int mt = job >> 1, nt = job & 1;
      v8f acc = {};
      for (int k0 = 0; k0 < LP; k0 += 32) {
        v16h a = frag_row(ph, LP, mt * 16, k0);
        v16h b = frag_row(vt, LP, nt * 16, k0);
        acc = wmma_f16(a, b, acc);
      }
      int n = nt * 16 + (lane & 15);
      int g = lane >> 4;
      if (n < DKK) {
#pragma unroll
        for (int j = 0; j < 8; ++j)
          y[(mt * 16 + j + g * 8) * YS + hd * DKK + n] = (_Float16)acc[j];
      }
    }
    __syncthreads();
  }

  // --- additive pool: s_l = tanh(y W1 + b1) . W2 ; softmax ; out = sum a_l y_l
  for (int l = tid; l < LP; l += 256) s[l] = 0.0f;
  __syncthreads();
  for (int job = wave; job < MT * POOL_NT; job += 8) {
    int mt = job / POOL_NT, nt = job - mt * POOL_NT;
    v8f acc = {};
    for (int kb = 0; kb < POOL_KB; ++kb) {
      v16h a = frag_row(y, YS, mt * 16, kb * 32);
      v16h b = frag_swz(W1h, POOL_NT, kb, nt);
      acc = wmma_f16(a, b, acc);
    }
    int n = nt * 16 + (lane & 15);
    int g = lane >> 4;
    float w2 = pW2[n];
    float b1 = pB1[n];
#pragma unroll
    for (int j = 0; j < 8; ++j) {
      int m = mt * 16 + j + g * 8;
      if (m < L) atomicAdd(&s[m], tanhf(acc[j] + b1) * w2);
    }
  }
  __syncthreads();
  if (tid == 0) {
    float mx = -1e30f;
    for (int l = 0; l < L; ++l) mx = fmaxf(mx, s[l]);
    float sum = 0.0f;
    for (int l = 0; l < L; ++l) {
      float e = __expf(s[l] - mx);
      s[l] = e;
      sum += e;
    }
    float inv = 1.0f / sum;
    for (int l = 0; l < L; ++l) s[l] *= inv;
  }
  __syncthreads();
  for (int d = tid; d < DD; d += 256) {
    float acc = 0.0f;
    for (int l = 0; l < L; ++l) acc += s[l] * (float)y[l * YS + d];
    outp[(size_t)item * DD + d] = acc;
  }
}

// view pool over {title, body, vert} rows + FC 400->64
__global__ __launch_bounds__(256) void qp_fc_kernel(
    const float* __restrict__ t_pool, const float* __restrict__ b_pool,
    const float* __restrict__ vemb, const int* __restrict__ news_vert,
    const int* __restrict__ user_vert, const float* __restrict__ qpW1,
    const float* __restrict__ qpB1, const float* __restrict__ qpW2,
    const float* __restrict__ fcW, const float* __restrict__ fcb,
    float* __restrict__ enc_out) {
  __shared__ float r[3][DD];
  __shared__ float sj[3];
  __shared__ float nv[DD];
  int item = blockIdx.x, tid = threadIdx.x;
  int vidx = (item < NNEWS) ? news_vert[item] : user_vert[item - NNEWS];
  for (int d = tid; d < DD; d += 256) {
    r[0][d] = t_pool[(size_t)item * DD + d];
    r[1][d] = b_pool[(size_t)item * DD + d];
    r[2][d] = vemb[(size_t)vidx * DD + d];
  }
  if (tid < 3) sj[tid] = 0.0f;
  __syncthreads();
  for (int idx = tid; idx < 3 * DD; idx += 256) {
    int j = idx / DD, k = idx - j * DD;
    float acc = qpB1[k];
    for (int d = 0; d < DD; ++d) acc += r[j][d] * qpW1[d * DD + k];
    atomicAdd(&sj[j], tanhf(acc) * qpW2[k]);
  }
  __syncthreads();
  if (tid == 0) {
    float mx = fmaxf(sj[0], fmaxf(sj[1], sj[2]));
    float e0 = __expf(sj[0] - mx), e1 = __expf(sj[1] - mx),
          e2 = __expf(sj[2] - mx);
    float inv = 1.0f / (e0 + e1 + e2);
    sj[0] = e0 * inv;
    sj[1] = e1 * inv;
    sj[2] = e2 * inv;
  }
  __syncthreads();
  for (int d = tid; d < DD; d += 256)
    nv[d] = sj[0] * r[0][d] + sj[1] * r[1][d] + sj[2] * r[2][d];
  __syncthreads();
  for (int o = tid; o < DOUT; o += 256) {
    float acc = fcb[o];
    for (int d = 0; d < DD; ++d) acc += nv[d] * fcW[d * DOUT + o];
    enc_out[(size_t)item * DOUT + o] = acc;
  }
}

// user-history pool (64-d) + final 16x5 score dots
__global__ __launch_bounds__(256) void final_kernel(
    const float* __restrict__ enc_out, const float* __restrict__ upW1,
    const float* __restrict__ upB1, const float* __restrict__ upW2,
    float* __restrict__ out) {
  __shared__ float u[HIST][DOUT];
  __shared__ float sl[HIST];
  __shared__ float uv[DOUT];
  int b = blockIdx.x, tid = threadIdx.x;
  for (int idx = tid; idx < HIST * DOUT; idx += 256) {
    int l = idx / DOUT, d = idx - l * DOUT;
    u[l][d] = enc_out[(size_t)(NNEWS + b * HIST + l) * DOUT + d];
  }
  if (tid < HIST) sl[tid] = 0.0f;
  __syncthreads();
  for (int idx = tid; idx < HIST * DOUT; idx += 256) {
    int l = idx / DOUT, k = idx - l * DOUT;
    float acc = upB1[k];
    for (int d = 0; d < DOUT; ++d) acc += u[l][d] * upW1[d * DOUT + k];
    atomicAdd(&sl[l], tanhf(acc) * upW2[k]);
  }
  __syncthreads();
  if (tid == 0) {
    float mx = -1e30f;
    for (int l = 0; l < HIST; ++l) mx = fmaxf(mx, sl[l]);
    float sum = 0.0f;
    for (int l = 0; l < HIST; ++l) {
      float e = __expf(sl[l] - mx);
      sl[l] = e;
      sum += e;
    }
    float inv = 1.0f / sum;
    for (int l = 0; l < HIST; ++l) sl[l] *= inv;
  }
  __syncthreads();
  for (int d = tid; d < DOUT; d += 256) {
    float acc = 0.0f;
    for (int l = 0; l < HIST; ++l) acc += sl[l] * u[l][d];
    uv[d] = acc;
  }
  __syncthreads();
  for (int n = tid; n < NPR; n += 256) {
    float acc = 0.0f;
    for (int d = 0; d < DOUT; ++d)
      acc += enc_out[(size_t)(b * NPR + n) * DOUT + d] * uv[d];
    out[b * NPR + n] = acc;
  }
}

extern "C" void kernel_launch(void* const* d_in, const int* in_sizes, int n_in,
                              void* d_out, int out_size, void* d_ws,
                              size_t ws_size, hipStream_t stream) {
  (void)in_sizes; (void)n_in; (void)out_size; (void)ws_size;
  const float* embed = (const float*)d_in[0];
  const float* vemb  = (const float*)d_in[1];
  const float* tWq = (const float*)d_in[2];
  const float* tbq = (const float*)d_in[3];
  const float* tWk = (const float*)d_in[4];
  const float* tbk = (const float*)d_in[5];
  const float* tWv = (const float*)d_in[6];
  const float* tbv = (const float*)d_in[7];
  const float* bWq = (const float*)d_in[8];
  const float* bbq = (const float*)d_in[9];
  const float* bWk = (const float*)d_in[10];
  const float* bbk = (const float*)d_in[11];
  const float* bWv = (const float*)d_in[12];
  const float* bbv = (const float*)d_in[13];
  const float* tpW1 = (const float*)d_in[14];
  const float* tpB1 = (const float*)d_in[15];
  const float* tpW2 = (const float*)d_in[16];
  const float* bpW1 = (const float*)d_in[18];
  const float* bpB1 = (const float*)d_in[19];
  const float* bpW2 = (const float*)d_in[20];
  const float* qpW1 = (const float*)d_in[22];
  const float* qpB1 = (const float*)d_in[23];
  const float* qpW2 = (const float*)d_in[24];
  const float* upW1 = (const float*)d_in[26];
  const float* upB1 = (const float*)d_in[27];
  const float* upW2 = (const float*)d_in[28];
  const float* fcW  = (const float*)d_in[30];
  const float* fcb  = (const float*)d_in[31];
  const int* user_title = (const int*)d_in[32];
  const int* user_vert  = (const int*)d_in[33];
  const int* user_body  = (const int*)d_in[34];
  const int* news_title = (const int*)d_in[35];
  const int* news_vert  = (const int*)d_in[36];
  const int* news_body  = (const int*)d_in[37];
  float* out = (float*)d_out;

  char* ws = (char*)d_ws;
  size_t off = 0;
  auto allocH = [&](size_t n) {
    _Float16* p = (_Float16*)(ws + off);
    off += (n * sizeof(_Float16) + 255) & ~(size_t)255;
    return p;
  };
  auto allocF = [&](size_t n) {
    float* p = (float*)(ws + off);
    off += (n * sizeof(float) + 255) & ~(size_t)255;
    return p;
  };
  const int nqkv = QKV_KB * QKV_NT * 512;    // 204800 swizzled halves
  const int npl  = POOL_KB * POOL_NT * 512;  // 166400 swizzled halves
  _Float16* tWqh = allocH(nqkv);
  _Float16* tWkh = allocH(nqkv);
  _Float16* tWvh = allocH(nqkv);
  _Float16* bWqh = allocH(nqkv);
  _Float16* bWkh = allocH(nqkv);
  _Float16* bWvh = allocH(nqkv);
  _Float16* tpW1h = allocH(npl);
  _Float16* bpW1h = allocH(npl);
  float* tbq_p = allocF(NPAD);
  float* tbk_p = allocF(NPAD);
  float* tbv_p = allocF(NPAD);
  float* bbq_p = allocF(NPAD);
  float* bbk_p = allocF(NPAD);
  float* bbv_p = allocF(NPAD);
  float* t_pool  = allocF((size_t)NITEMS * DD);
  float* b_pool  = allocF((size_t)NITEMS * DD);
  float* enc_out = allocF((size_t)NITEMS * DOUT);

  prep_qkv_swz<<<(nqkv + 255) / 256, 256, 0, stream>>>(tWq, tWqh, nqkv);
  prep_qkv_swz<<<(nqkv + 255) / 256, 256, 0, stream>>>(tWk, tWkh, nqkv);
  prep_qkv_swz<<<(nqkv + 255) / 256, 256, 0, stream>>>(tWv, tWvh, nqkv);
  prep_qkv_swz<<<(nqkv + 255) / 256, 256, 0, stream>>>(bWq, bWqh, nqkv);
  prep_qkv_swz<<<(nqkv + 255) / 256, 256, 0, stream>>>(bWk, bWkh, nqkv);
  prep_qkv_swz<<<(nqkv + 255) / 256, 256, 0, stream>>>(bWv, bWvh, nqkv);
  prep_pool_swz<<<(npl + 255) / 256, 256, 0, stream>>>(tpW1, tpW1h, npl);
  prep_pool_swz<<<(npl + 255) / 256, 256, 0, stream>>>(bpW1, bpW1h, npl);
  prep_bias_pad<<<3, 256, 0, stream>>>(tbq, tbq_p);
  prep_bias_pad<<<3, 256, 0, stream>>>(tbk, tbk_p);
  prep_bias_pad<<<3, 256, 0, stream>>>(tbv, tbv_p);
  prep_bias_pad<<<3, 256, 0, stream>>>(bbq, bbq_p);
  prep_bias_pad<<<3, 256, 0, stream>>>(bbk, bbk_p);
  prep_bias_pad<<<3, 256, 0, stream>>>(bbv, bbv_p);

  auto shbytes = [](size_t LP) {
    return LP * (EKS + YS + 96) * 2 + LP * LP * 6 + LP * 4;
  };
  encoder_kernel<TLEN, 32><<<NITEMS, 256, shbytes(32), stream>>>(
      news_title, user_title, embed, tWqh, tWkh, tWvh, tbq_p, tbk_p, tbv_p,
      tpW1h, tpB1, tpW2, t_pool);
  encoder_kernel<BLEN, 128><<<NITEMS, 256, shbytes(128), stream>>>(
      news_body, user_body, embed, bWqh, bWkh, bWvh, bbq_p, bbk_p, bbv_p,
      bpW1h, bpB1, bpW2, b_pool);
  qp_fc_kernel<<<NITEMS, 256, 0, stream>>>(t_pool, b_pool, vemb, news_vert,
                                           user_vert, qpW1, qpB1, qpW2, fcW,
                                           fcb, enc_out);
  final_kernel<<<BZd, 256, 0, stream>>>(enc_out, upW1, upB1, upW2, out);
}
